// Block_26044681683324
// MI455X (gfx1250) — matrix operand order repaired and checked by reading, not compiled, not run on hardware
//
#include <hip/hip_runtime.h>
#include <hip/hip_bf16.h>
#include <math.h>
#include <stdint.h>

typedef __attribute__((ext_vector_type(16))) _Float16 v16h;
typedef __attribute__((ext_vector_type(8)))  float    v8f;

union FragH { v16h h; uint4 u[2]; };

__device__ __forceinline__ v8f wmma_f16(v16h a, v16h b, v8f c) {
    // D = A(16x32 f16) * B(32x16 f16) + C(16x16 f32)
    return __builtin_amdgcn_wmma_f32_16x16x32_f16(
        /*neg_a=*/false, a, /*neg_b=*/false, b,
        /*c_mod=*/(short)0, c, /*reuse_a=*/false, /*reuse_b=*/false);
}

#define NB 768
#define TT 2048
#define NHEAD 12

// ---------------------------------------------------------------------------
// Async global->LDS staging (CDNA5 GLOBAL_LOAD_ASYNC_TO_LDS_B128, ASYNCcnt).
// Builtin signature (from clang diagnostic): param0 = int vec4 AS(1)*,
// param1 = int vec4 AS(3)*, then two immediates (offset, cpol).
// AS(3) pointer built by integer truncation: low 32 bits of a flat LDS address
// are the LDS byte offset (ISA 10.2 aperture rule); AS(3) pointers are 32-bit.
// ---------------------------------------------------------------------------
#if __has_builtin(__builtin_amdgcn_global_load_async_to_lds_b128)
#define HAS_ASYNC 1
#else
#define HAS_ASYNC 0
#endif

typedef int vec4i __attribute__((vector_size(16)));
typedef __attribute__((address_space(1))) vec4i* g128p;
typedef __attribute__((address_space(3))) vec4i* l128p;

#if HAS_ASYNC
#if __has_builtin(__builtin_amdgcn_s_wait_asynccnt)
#define WAIT_ASYNC(n) __builtin_amdgcn_s_wait_asynccnt(n)
#else
#define WAIT_ASYNC(n) asm volatile("s_wait_asynccnt %0" ::"i"(n) : "memory")
#endif
#else
#define WAIT_ASYNC(n) ((void)0)
#endif

__device__ __forceinline__ void cp_b128(_Float16* l, const _Float16* g) {
#if HAS_ASYNC
    __builtin_amdgcn_global_load_async_to_lds_b128(
        (g128p)(unsigned long long)(uintptr_t)g,
        (l128p)(unsigned)(uintptr_t)l, 0, 0);
#else
    *(uint4*)l = *(const uint4*)g;
#endif
}

// ---------------------------------------------------------------------------
// Weight convert: W[K][N] fp32 -> Wt[N][K] f16 (transposed for contiguous B tiles)
// ---------------------------------------------------------------------------
__global__ void wt_convert(const float* __restrict__ W, _Float16* __restrict__ Wt,
                           int K, int N) {
    size_t idx = (size_t)blockIdx.x * 256 + threadIdx.x;
    size_t total = (size_t)K * N;
    if (idx >= total) return;
    int n = (int)(idx / K);
    int k = (int)(idx % K);
    Wt[idx] = (_Float16)W[(size_t)k * N + n];
}

// ---------------------------------------------------------------------------
// LayerNorm: one 768-wide row per 256-thread block, output f16
// ---------------------------------------------------------------------------
__global__ __launch_bounds__(256)
void ln_kernel(const float* __restrict__ x, const float* __restrict__ w,
               const float* __restrict__ b, _Float16* __restrict__ out) {
    const int row = blockIdx.x;
    const int tid = threadIdx.x;
    const float* xr = x + (size_t)row * NB;
    float v0 = xr[tid], v1 = xr[tid + 256], v2 = xr[tid + 512];
    float s = v0 + v1 + v2;
    for (int msk = 16; msk; msk >>= 1) s += __shfl_xor(s, msk, 32);
    __shared__ float red[8];
    __shared__ float red2[8];
    const int wv = tid >> 5, ln = tid & 31;
    if (ln == 0) red[wv] = s;
    __syncthreads();
    float tot = 0.f;
#pragma unroll
    for (int i = 0; i < 8; i++) tot += red[i];
    const float mean = tot * (1.0f / 768.0f);
    const float d0 = v0 - mean, d1 = v1 - mean, d2 = v2 - mean;
    float q = d0 * d0 + d1 * d1 + d2 * d2;
    for (int msk = 16; msk; msk >>= 1) q += __shfl_xor(q, msk, 32);
    if (ln == 0) red2[wv] = q;
    __syncthreads();
    float tq = 0.f;
#pragma unroll
    for (int i = 0; i < 8; i++) tq += red2[i];
    const float stdv = sqrtf(tq * (1.0f / 768.0f));
    const float inv = 1.0f / (stdv + 1e-6f);
    _Float16* o = out + (size_t)row * NB;
    o[tid]       = (_Float16)(w[tid]       * d0 * inv + b[tid]);
    o[tid + 256] = (_Float16)(w[tid + 256] * d1 * inv + b[tid + 256]);
    o[tid + 512] = (_Float16)(w[tid + 512] * d2 * inv + b[tid + 512]);
}

// ---------------------------------------------------------------------------
// WMMA GEMM: C[M,N] = A[M,K](f16) * Wt[N,K]^T(f16) + bias, tile 128x128x32,
// double-buffered LDS with async global->LDS staging.
// 8 waves: wave w -> rows (w&3)*32, cols (w>>2)*64 => 2x4 WMMA frags.
// MODE 0: split into q/k/v [B,H,T,D] f16 (q scaled 1/8)
// MODE 1: fp32 out = acc + bias + resid
// MODE 2: f16 out = gelu(acc + bias)
// ---------------------------------------------------------------------------
template <int MODE>
__global__ __launch_bounds__(256)
void gemm_kernel(const _Float16* __restrict__ A, const _Float16* __restrict__ Bt,
                 const float* __restrict__ bias, int K, int N,
                 const float* __restrict__ resid, float* __restrict__ outf,
                 _Float16* __restrict__ outh,
                 _Float16* __restrict__ qo, _Float16* __restrict__ ko,
                 _Float16* __restrict__ vo) {
    __shared__ __align__(16) _Float16 sA[2][128 * 32];
    __shared__ __align__(16) _Float16 sB[2][128 * 32];
    const int tid = threadIdx.x;
    const int w = tid >> 5, lane = tid & 31;
    const int wr = w & 3, wc = w >> 2;
    const int ln16 = lane & 15, hi = lane >> 4;
    const int mb0 = blockIdx.y * 128, nb0 = blockIdx.x * 128;

    v8f acc[2][4];
    const v8f zero = {0.f, 0.f, 0.f, 0.f, 0.f, 0.f, 0.f, 0.f};
#pragma unroll
    for (int i = 0; i < 2; i++)
#pragma unroll
        for (int j = 0; j < 4; j++) acc[i][j] = zero;

    // 4 async b128 transfers per thread per tile (2 for A, 2 for B)
    auto stage = [&](int buf, int kb) {
        for (int c = tid; c < 512; c += 256) {
            const int row = c >> 2, ck = c & 3;
            cp_b128(&sA[buf][row * 32 + ck * 8],
                    A + (size_t)(mb0 + row) * K + kb + ck * 8);
            cp_b128(&sB[buf][row * 32 + ck * 8],
                    Bt + (size_t)(nb0 + row) * K + kb + ck * 8);
        }
    };

    const int nk = K >> 5;
    stage(0, 0);
    for (int ik = 0; ik < nk; ik++) {
        const int cur = ik & 1;
        if (ik + 1 < nk) {
            stage(cur ^ 1, (ik + 1) * 32);
            WAIT_ASYNC(4);  // in-order: current tile's 4 transfers complete
        } else {
            WAIT_ASYNC(0);
        }
        __syncthreads();  // current buffer visible to all waves

        FragH a[2];
#pragma unroll
        for (int i = 0; i < 2; i++) {
            const int ro = (wr * 32 + i * 16 + ln16) * 64;
            a[i].u[0] = *(const uint4*)((const char*)sA[cur] + ro + hi * 16);
            a[i].u[1] = *(const uint4*)((const char*)sA[cur] + ro + 32 + hi * 16);
        }
#pragma unroll
        for (int j = 0; j < 4; j++) {
            FragH bf;
            const int ro = (wc * 64 + j * 16 + ln16) * 64;
            bf.u[0] = *(const uint4*)((const char*)sB[cur] + ro + hi * 16);
            bf.u[1] = *(const uint4*)((const char*)sB[cur] + ro + 32 + hi * 16);
#pragma unroll
            for (int i = 0; i < 2; i++)
                acc[i][j] = wmma_f16(a[i].h, bf.h, acc[i][j]);
        }
        __syncthreads();  // all reads of 'cur' done before it is restaged
    }

    // Epilogue. C layout: lane holds col N=ln16, rows M = hi*8 + r.
#pragma unroll
    for (int i = 0; i < 2; i++) {
#pragma unroll
        for (int j = 0; j < 4; j++) {
            const int col = nb0 + wc * 64 + j * 16 + ln16;
            const float bsv = bias[col];
#pragma unroll
            for (int r = 0; r < 8; r++) {
                const int rowg = mb0 + wr * 32 + i * 16 + hi * 8 + r;
                const float val = acc[i][j][r] + bsv;
                if (MODE == 0) {
                    const int sel = col / NB, cc = col % NB;
                    const int h = cc >> 6, d = cc & 63;
                    const int bb = rowg >> 11, t = rowg & (TT - 1);
                    const size_t o = (((size_t)bb * NHEAD + h) * TT + t) * 64 + d;
                    if (sel == 0)       qo[o] = (_Float16)(val * 0.125f);
                    else if (sel == 1)  ko[o] = (_Float16)val;
                    else                vo[o] = (_Float16)val;
                } else if (MODE == 1) {
                    const size_t o = (size_t)rowg * N + col;
                    outf[o] = val + resid[o];
                } else {
                    const float g = 0.5f * val * (1.0f + erff(val * 0.70710678118654752f));
                    outh[(size_t)rowg * N + col] = (_Float16)g;
                }
            }
        }
    }
}

// ---------------------------------------------------------------------------
// Flash attention (causal). grid = (T/128 q-tiles, B*H). 8 waves, wave w owns
// q-rows [w*16, w*16+16) of the 128-row tile; online softmax over s-tiles of 64.
// ---------------------------------------------------------------------------
__global__ __launch_bounds__(256)
void attn_kernel(const _Float16* __restrict__ Q, const _Float16* __restrict__ Kb,
                 const _Float16* __restrict__ Vb, _Float16* __restrict__ Y) {
    __shared__ __align__(16) _Float16 sQ[128 * 64];
    __shared__ __align__(16) _Float16 sK[64 * 64];
    __shared__ __align__(16) _Float16 sVt[64 * 64];  // transposed: [d][s]
    __shared__ __align__(16) _Float16 sP[128 * 64];
    const int tid = threadIdx.x, w = tid >> 5, lane = tid & 31;
    const int qt = blockIdx.x, bh = blockIdx.y;
    const int bidx = bh / NHEAD, h = bh % NHEAD;
    const int qbase = qt * 128;
    const int ln16 = lane & 15, hi = lane >> 4;

    const _Float16* qsrc = Q + ((size_t)bh * TT + qbase) * 64;
    for (int c = tid; c < 1024; c += 256) {  // 128 rows * 8 chunks of 16B
        const int row = c >> 3, ck = c & 7;
        cp_b128(&sQ[row * 64 + ck * 8], qsrc + (size_t)row * 64 + ck * 8);
    }

    const v8f zero = {0.f, 0.f, 0.f, 0.f, 0.f, 0.f, 0.f, 0.f};
    v8f o[4];
#pragma unroll
    for (int j = 0; j < 4; j++) o[j] = zero;
    float mst[8], lst[8];
#pragma unroll
    for (int r = 0; r < 8; r++) { mst[r] = -1e30f; lst[r] = 0.f; }

    const int nst = 2 * qt + 2;  // s-tiles needed for causal coverage
    for (int st = 0; st < nst; ++st) {
        __syncthreads();  // previous tile's sK/sVt reads complete
        const _Float16* ksrc = Kb + ((size_t)bh * TT + st * 64) * 64;
        const _Float16* vsrc = Vb + ((size_t)bh * TT + st * 64) * 64;
        for (int c = tid; c < 512; c += 256) {  // 64 rows * 8 chunks
            const int row = c >> 3, ck = c & 7;
            cp_b128(&sK[row * 64 + ck * 8], ksrc + (size_t)row * 64 + ck * 8);
            FragH tmp;
            tmp.u[0] = *(const uint4*)(vsrc + (size_t)row * 64 + ck * 8);
#pragma unroll
            for (int e = 0; e < 8; e++)
                sVt[(ck * 8 + e) * 64 + row] = tmp.h[e];
        }
        WAIT_ASYNC(0);
        __syncthreads();

        // S = Q * K^T  (A frags from sQ, B frags: B[k=d][n=s] = K[s][d] = sK rows)
        FragH qa[2];
#pragma unroll
        for (int ka = 0; ka < 2; ka++) {
            const int ro = (w * 16 + ln16) * 128 + ka * 64;
            qa[ka].u[0] = *(const uint4*)((const char*)sQ + ro + hi * 16);
            qa[ka].u[1] = *(const uint4*)((const char*)sQ + ro + 32 + hi * 16);
        }
        v8f sfr[4];
#pragma unroll
        for (int j = 0; j < 4; j++) {
            v8f cacc = zero;
#pragma unroll
            for (int ka = 0; ka < 2; ka++) {
                FragH kf;
                const int ro = (j * 16 + ln16) * 128 + ka * 64;
                kf.u[0] = *(const uint4*)((const char*)sK + ro + hi * 16);
                kf.u[1] = *(const uint4*)((const char*)sK + ro + 32 + hi * 16);
                cacc = wmma_f16(qa[ka].h, kf.h, cacc);
            }
            sfr[j] = cacc;
        }

        // causal mask only on diagonal-straddling tiles (wave-uniform skip);
        // q pre-scaled by 1/sqrt(D) at QKV epilogue.
        if (st * 64 + 63 > qbase + w * 16) {
#pragma unroll
            for (int j = 0; j < 4; j++) {
                const int sg = st * 64 + j * 16 + ln16;
#pragma unroll
                for (int r = 0; r < 8; r++) {
                    const int qg = qbase + w * 16 + hi * 8 + r;
                    if (sg > qg) sfr[j][r] = -1e30f;
                }
            }
        }

        // online softmax: rows live in 16-lane halves; xor masks 1/2/4/8 reduce cols
#pragma unroll
        for (int r = 0; r < 8; r++) {
            float t = fmaxf(fmaxf(sfr[0][r], sfr[1][r]), fmaxf(sfr[2][r], sfr[3][r]));
            t = fmaxf(t, __shfl_xor(t, 1, 32));
            t = fmaxf(t, __shfl_xor(t, 2, 32));
            t = fmaxf(t, __shfl_xor(t, 4, 32));
            t = fmaxf(t, __shfl_xor(t, 8, 32));
            const float mn = fmaxf(mst[r], t);
            const float sc = __expf(mst[r] - mn);
            float rs = 0.f;
#pragma unroll
            for (int j = 0; j < 4; j++) {
                const float p = __expf(sfr[j][r] - mn);
                sfr[j][r] = p;
                rs += p;
            }
            rs += __shfl_xor(rs, 1, 32);
            rs += __shfl_xor(rs, 2, 32);
            rs += __shfl_xor(rs, 4, 32);
            rs += __shfl_xor(rs, 8, 32);
            lst[r] = lst[r] * sc + rs;
            mst[r] = mn;
#pragma unroll
            for (int j = 0; j < 4; j++) o[j][r] *= sc;
        }

        // round-trip P through LDS (C layout -> A layout), per-wave private region
#pragma unroll
        for (int j = 0; j < 4; j++)
#pragma unroll
            for (int r = 0; r < 8; r++)
                sP[(w * 16 + hi * 8 + r) * 64 + j * 16 + ln16] = (_Float16)sfr[j][r];

        FragH pa[2];
#pragma unroll
        for (int ka = 0; ka < 2; ka++) {
            const int ro = (w * 16 + ln16) * 128 + ka * 64;
            pa[ka].u[0] = *(const uint4*)((const char*)sP + ro + hi * 16);
            pa[ka].u[1] = *(const uint4*)((const char*)sP + ro + 32 + hi * 16);
        }
        // O += P * V   (B[k=s][n=d] = V[s][d] = sVt[d][s] rows)
#pragma unroll
        for (int j = 0; j < 4; j++) {
#pragma unroll
            for (int ka = 0; ka < 2; ka++) {
                FragH vf;
                const int ro = (j * 16 + ln16) * 128 + ka * 64;
                vf.u[0] = *(const uint4*)((const char*)sVt + ro + hi * 16);
                vf.u[1] = *(const uint4*)((const char*)sVt + ro + 32 + hi * 16);
                o[j] = wmma_f16(pa[ka].h, vf.h, o[j]);
            }
        }
    }

    // normalize + write y[b, t, h*64 + d] as f16
#pragma unroll
    for (int r = 0; r < 8; r++) {
        const float inv = 1.0f / lst[r];
        const int rowg = qbase + w * 16 + hi * 8 + r;
        const size_t base = ((size_t)bidx * TT + rowg) * NB + h * 64;
#pragma unroll
        for (int j = 0; j < 4; j++)
            Y[base + j * 16 + ln16] = (_Float16)(o[j][r] * inv);
    }
}

// ---------------------------------------------------------------------------
extern "C" void kernel_launch(void* const* d_in, const int* in_sizes, int n_in,
                              void* d_out, int out_size, void* d_ws, size_t ws_size,
                              hipStream_t stream) {
    (void)in_sizes; (void)n_in; (void)out_size; (void)ws_size;
    const float* x     = (const float*)d_in[0];
    const float* ln1_w = (const float*)d_in[1];
    const float* ln1_b = (const float*)d_in[2];
    const float* W_attn = (const float*)d_in[3];
    const float* b_attn = (const float*)d_in[4];
    const float* W_ap  = (const float*)d_in[5];
    const float* b_ap  = (const float*)d_in[6];
    const float* ln2_w = (const float*)d_in[7];
    const float* ln2_b = (const float*)d_in[8];
    const float* W_fc  = (const float*)d_in[9];
    const float* b_fc  = (const float*)d_in[10];
    const float* W_mp  = (const float*)d_in[11];
    const float* b_mp  = (const float*)d_in[12];

    const int M = 4 * 2048;  // 8192 tokens

    char* p = (char*)d_ws;
    auto alloc = [&](size_t bytes) -> void* {
        void* r = (void*)p;
        p += (bytes + 255) & ~(size_t)255;
        return r;
    };
    _Float16* wt_attn = (_Float16*)alloc((size_t)2304 * 768 * 2);
    _Float16* wt_proj = (_Float16*)alloc((size_t)768 * 768 * 2);
    _Float16* wt_fc   = (_Float16*)alloc((size_t)3072 * 768 * 2);
    _Float16* wt_mp   = (_Float16*)alloc((size_t)768 * 3072 * 2);
    _Float16* h1      = (_Float16*)alloc((size_t)M * 768 * 2);
    _Float16* qb      = (_Float16*)alloc((size_t)M * 768 * 2);
    _Float16* kb      = (_Float16*)alloc((size_t)M * 768 * 2);
    _Float16* vb      = (_Float16*)alloc((size_t)M * 768 * 2);
    _Float16* yb      = (_Float16*)alloc((size_t)M * 768 * 2);
    float*    x1      = (float*)   alloc((size_t)M * 768 * 4);
    _Float16* h2      = (_Float16*)alloc((size_t)M * 768 * 2);
    _Float16* afc     = (_Float16*)alloc((size_t)M * 3072 * 2);

    // weight conversion (transposed, f16)
    wt_convert<<<(2304 * 768 + 255) / 256, 256, 0, stream>>>(W_attn, wt_attn, 768, 2304);
    wt_convert<<<(768 * 768 + 255) / 256, 256, 0, stream>>>(W_ap, wt_proj, 768, 768);
    wt_convert<<<(3072 * 768 + 255) / 256, 256, 0, stream>>>(W_fc, wt_fc, 768, 3072);
    wt_convert<<<(768 * 3072 + 255) / 256, 256, 0, stream>>>(W_mp, wt_mp, 3072, 768);

    // h1 = LN1(x)
    ln_kernel<<<M, 256, 0, stream>>>(x, ln1_w, ln1_b, h1);

    // qkv = h1 @ W_attn + b_attn  -> split q/k/v [B,H,T,D]
    gemm_kernel<0><<<dim3(2304 / 128, M / 128), 256, 0, stream>>>(
        h1, wt_attn, b_attn, 768, 2304, nullptr, nullptr, nullptr, qb, kb, vb);

    // y = causal attention
    attn_kernel<<<dim3(2048 / 128, 4 * NHEAD), 256, 0, stream>>>(qb, kb, vb, yb);

    // x1 = x + y @ W_attn_proj + b_attn_proj
    gemm_kernel<1><<<dim3(768 / 128, M / 128), 256, 0, stream>>>(
        yb, wt_proj, b_ap, 768, 768, x, x1, nullptr, nullptr, nullptr, nullptr);

    // h2 = LN2(x1)
    ln_kernel<<<M, 256, 0, stream>>>(x1, ln2_w, ln2_b, h2);

    // afc = gelu(h2 @ W_fc + b_fc)
    gemm_kernel<2><<<dim3(3072 / 128, M / 128), 256, 0, stream>>>(
        h2, wt_fc, b_fc, 768, 3072, nullptr, nullptr, afc, nullptr, nullptr, nullptr);

    // out = x1 + afc @ W_mlp_proj + b_mlp_proj
    gemm_kernel<1><<<dim3(768 / 128, M / 128), 256, 0, stream>>>(
        afc, wt_mp, b_mp, 3072, 768, x1, (float*)d_out, nullptr, nullptr, nullptr, nullptr);
}
